// PABasicBlock_6983616824332
// MI455X (gfx1250) — matrix operand order repaired and checked by reading, not compiled, not run on hardware
//
#include <hip/hip_runtime.h>
#include <hip/hip_bf16.h>

typedef __attribute__((ext_vector_type(16))) _Float16 v16h;
typedef __attribute__((ext_vector_type(8)))  _Float16 v8h;
typedef __attribute__((ext_vector_type(8)))  float    v8f;

#define BQ 64
#define CQ 128
#define HQ 56
#define WQ 56
#define KSTEPS 36           // 9 taps * (128/32) channel blocks
#define PACK_ELEMS (KSTEPS * CQ * 32)   // 147456 f16 per conv

// ---------------------------------------------------------------------------
// BN constant folding: s = gamma/sqrt(var+eps), t = beta - mean*s
// ---------------------------------------------------------------------------
__global__ void prep_consts(const float* __restrict__ g1, const float* __restrict__ be1,
                            const float* __restrict__ m1, const float* __restrict__ v1,
                            const float* __restrict__ g2, const float* __restrict__ be2,
                            const float* __restrict__ m2, const float* __restrict__ v2,
                            float* __restrict__ s1, float* __restrict__ t1,
                            float* __restrict__ s2, float* __restrict__ t2) {
    int c = threadIdx.x;
    if (c < CQ) {
        float a = g1[c] * rsqrtf(v1[c] + 1e-5f);
        s1[c] = a; t1[c] = be1[c] - m1[c] * a;
        float d = g2[c] * rsqrtf(v2[c] + 1e-5f);
        s2[c] = d; t2[c] = be2[c] - m2[c] * d;
    }
}

// ---------------------------------------------------------------------------
// Binarize weights (sign, sign(0)=0) and repack to [kstep][n=co][k_in_32] f16
// kstep s: tap = s>>2 (kh=tap/3, kw=tap%3), ci = (s&3)*32 + kk
// ---------------------------------------------------------------------------
__global__ void prep_weights(const float* __restrict__ w1, const float* __restrict__ w2,
                             _Float16* __restrict__ p1, _Float16* __restrict__ p2) {
    int idx = blockIdx.x * 256 + threadIdx.x;
    if (idx >= 2 * PACK_ELEMS) return;
    int conv = idx / PACK_ELEMS;
    int r    = idx % PACK_ELEMS;
    int kk = r & 31;
    int n  = (r >> 5) & (CQ - 1);
    int s  = r >> 12;                 // /(128*32)
    int tap = s >> 2;
    int kh = tap / 3, kw = tap % 3;
    int ci = (s & 3) * 32 + kk;
    const float* w = conv ? w2 : w1;
    float v = w[((n * CQ + ci) * 3 + kh) * 3 + kw];
    float sv = (v > 0.f) ? 1.f : ((v < 0.f) ? -1.f : 0.f);
    (conv ? p2 : p1)[r] = (_Float16)sv;
}

// ---------------------------------------------------------------------------
// x (f32 NCHW) -> xt (f16 NHWC) with bn1 fused. LDS-tiled transpose, one
// block per (b,h): read coalesced along w, write coalesced along c.
// ---------------------------------------------------------------------------
__global__ __launch_bounds__(256)
void transform_kernel(const float* __restrict__ x, const float* __restrict__ s1,
                      const float* __restrict__ t1, _Float16* __restrict__ xt) {
    __shared__ float tile[CQ * WQ];   // 28 KB
    const int h = blockIdx.x, b = blockIdx.y, tid = threadIdx.x;
    for (int i = tid; i < CQ * WQ; i += 256) {
        int c = i / WQ, w = i % WQ;
        tile[i] = x[(long)(b * CQ + c) * (HQ * WQ) + h * WQ + w];
    }
    __syncthreads();
    _Float16* dst = xt + (long)((b * HQ + h) * WQ) * CQ;
    for (int i = tid; i < CQ * WQ; i += 256) {
        int w = i / CQ, c = i % CQ;
        float v = tile[c * WQ + w] * s1[c] + t1[c];
        dst[w * CQ + c] = (_Float16)v;
    }
}

// ---------------------------------------------------------------------------
// Implicit-GEMM binarized 3x3 conv via v_wmma_f32_16x16x32_f16.
// One block per (b,h): M-tile = 64 (row of w, padded), N = 128, K = 1152.
// 8 waves: wave = msub(0..3) x ngrp(0..1); each wave owns 1 M-subtile x 4
// N-subtiles (4 v8f accumulators), 36 K-steps.
// MODE 1: out16 = f16((acc + bias)*s2 + t2)   (NHWC f16, feeds conv2)
// MODE 2: out32 = acc + bias + residual       (NCHW f32, final output)
// ---------------------------------------------------------------------------
template <int MODE>
__global__ __launch_bounds__(256)
void conv_kernel(const _Float16* __restrict__ in,    // f16 NHWC [B][H][W][C]
                 const _Float16* __restrict__ bpack, // [36][128][32] f16
                 const float* __restrict__ bias,
                 const float* __restrict__ scale,    // MODE 1
                 const float* __restrict__ shift,    // MODE 1
                 const float* __restrict__ resid,    // MODE 2, f32 NCHW
                 _Float16* __restrict__ out16,       // MODE 1
                 float* __restrict__ out32) {        // MODE 2
    const int h = blockIdx.x;
    const int b = blockIdx.y;
    const int tid = threadIdx.x;

    // 3 input rows (h-1..h+1) x 66 w-slots (w' = -1..64, zero-padded) x 128 ch
    __shared__ _Float16 smemA[3 * 66 * CQ];   // 49.5 KB

    {   // cooperative staged load: 16 threads per 256B pixel-row, zero halo
        const int part = tid & 15;
        for (int row = tid >> 4; row < 3 * 66; row += 16) {
            int r = row / 66, wp = row % 66;
            int hin = h - 1 + r;
            int win = wp - 1;
            uint4 val = make_uint4(0u, 0u, 0u, 0u);
            if ((unsigned)hin < (unsigned)HQ && (unsigned)win < (unsigned)WQ) {
                val = *(const uint4*)(in + (long)(((b * HQ + hin) * WQ + win) * CQ) + part * 8);
            }
            *(uint4*)(&smemA[row * CQ + part * 8]) = val;
        }
    }
    __syncthreads();

    const int lane  = tid & 31;
    const int wave  = tid >> 5;
    const int msub  = wave & 3;     // M subtile (16 rows of w)
    const int ngrp  = wave >> 2;    // N half (cols 0..63 / 64..127)
    const int l16   = lane & 15;
    const int lhalf = lane >> 4;    // K-half selector per WMMA A/B layout
    const int m     = msub * 16 + l16;   // A-matrix row (output w) for this lane

    v8f acc0 = {}, acc1 = {}, acc2 = {}, acc3 = {};

    for (int s = 0; s < KSTEPS; ++s) {
        const int tap = s >> 2;
        const int dh = tap / 3, dw = tap % 3;
        const int c0 = (s & 3) * 32;

        // ---- A fragment from LDS (contiguous K order, same as B: lanes 0-15 K={0..15},
        //      lanes 16-31 K={16..31}) ----
        const _Float16* ap = &smemA[(dh * 66 + (m + dw)) * CQ + c0 + lhalf * 16];
        v8h alo = *(const v8h*)ap;
        v8h ahi = *(const v8h*)(ap + 8);
        v16h afrag;
#pragma unroll
        for (int i = 0; i < 8; ++i) { afrag[i] = alo[i]; afrag[8 + i] = ahi[i]; }

        // ---- B fragments from packed global (n fixed per lane, K contiguous) ----
        const _Float16* bb = bpack + (long)((s * CQ + ngrp * 64 + l16) * 32) + lhalf * 16;
        v16h bf0 = *(const v16h*)(bb + 0 * 512);
        v16h bf1 = *(const v16h*)(bb + 1 * 512);
        v16h bf2 = *(const v16h*)(bb + 2 * 512);
        v16h bf3 = *(const v16h*)(bb + 3 * 512);
        if (s + 1 < KSTEPS)
            __builtin_prefetch((const void*)(bb + CQ * 32), 0, 1);  // next K-step

        acc0 = __builtin_amdgcn_wmma_f32_16x16x32_f16(false, afrag, false, bf0,
                                                      (short)0, acc0, false, false);
        acc1 = __builtin_amdgcn_wmma_f32_16x16x32_f16(false, afrag, false, bf1,
                                                      (short)0, acc1, false, false);
        acc2 = __builtin_amdgcn_wmma_f32_16x16x32_f16(false, afrag, false, bf2,
                                                      (short)0, acc2, false, false);
        acc3 = __builtin_amdgcn_wmma_f32_16x16x32_f16(false, afrag, false, bf3,
                                                      (short)0, acc3, false, false);
    }

    // ---- epilogue: C/D layout — VGPR r holds M=r (lanes 0-15) / M=r+8 (16-31),
    //      N = l16 within subtile ----
    v8f accs[4] = {acc0, acc1, acc2, acc3};
#pragma unroll
    for (int j = 0; j < 4; ++j) {
        const int n = ngrp * 64 + j * 16 + l16;
        const float bs = bias[n];
        float sc = 0.f, sh = 0.f;
        if (MODE == 1) { sc = scale[n]; sh = shift[n]; }
#pragma unroll
        for (int r = 0; r < 8; ++r) {
            const int w = msub * 16 + r + lhalf * 8;
            if (w < WQ) {
                const float v = accs[j][r] + bs;
                if (MODE == 1) {
                    out16[(long)(((b * HQ + h) * WQ + w) * CQ) + n] = (_Float16)(v * sc + sh);
                } else {
                    const long oi = (long)((b * CQ + n) * HQ + h) * WQ + w;
                    out32[oi] = v + resid[oi];
                }
            }
        }
    }
}

// ---------------------------------------------------------------------------
extern "C" void kernel_launch(void* const* d_in, const int* in_sizes, int n_in,
                              void* d_out, int out_size, void* d_ws, size_t ws_size,
                              hipStream_t stream) {
    const float* x   = (const float*)d_in[0];
    const float* g1  = (const float*)d_in[1];
    const float* be1 = (const float*)d_in[2];
    const float* m1  = (const float*)d_in[3];
    const float* v1  = (const float*)d_in[4];
    const float* w1  = (const float*)d_in[5];
    const float* b1  = (const float*)d_in[6];
    const float* g2  = (const float*)d_in[7];
    const float* be2 = (const float*)d_in[8];
    const float* m2  = (const float*)d_in[9];
    const float* v2  = (const float*)d_in[10];
    const float* w2  = (const float*)d_in[11];
    const float* b2  = (const float*)d_in[12];

    char* ws = (char*)d_ws;
    const size_t ACT_BYTES = (size_t)BQ * HQ * WQ * CQ * sizeof(_Float16);  // 51,380,224
    _Float16* xt = (_Float16*)ws;
    _Float16* y1 = (_Float16*)(ws + ACT_BYTES);
    _Float16* p1 = (_Float16*)(ws + 2 * ACT_BYTES);
    _Float16* p2 = (_Float16*)(ws + 2 * ACT_BYTES + (size_t)PACK_ELEMS * 2);
    float*    s1 = (float*)(ws + 2 * ACT_BYTES + 2 * (size_t)PACK_ELEMS * 2);
    float*    t1 = s1 + CQ;
    float*    s2 = s1 + 2 * CQ;
    float*    t2 = s1 + 3 * CQ;

    prep_consts<<<1, 128, 0, stream>>>(g1, be1, m1, v1, g2, be2, m2, v2, s1, t1, s2, t2);
    prep_weights<<<(2 * PACK_ELEMS + 255) / 256, 256, 0, stream>>>(w1, w2, p1, p2);
    transform_kernel<<<dim3(HQ, BQ), 256, 0, stream>>>(x, s1, t1, xt);
    conv_kernel<1><<<dim3(HQ, BQ), 256, 0, stream>>>(xt, p1, b1, s2, t2, nullptr, y1, nullptr);
    conv_kernel<2><<<dim3(HQ, BQ), 256, 0, stream>>>(y1, p2, b2, nullptr, nullptr, x, nullptr,
                                                     (float*)d_out);
}
